// LSTM_28321014350129
// MI455X (gfx1250) — matrix operand order repaired and checked by reading, not compiled, not run on hardware
//
#include <hip/hip_runtime.h>
#include <hip/hip_bf16.h>

// ---------------- types for WMMA ----------------
typedef __attribute__((ext_vector_type(16))) __bf16 v16bf;
typedef __attribute__((ext_vector_type(8)))  __bf16 v8bf;
typedef __attribute__((ext_vector_type(8)))  float  v8f;
typedef int v4i __attribute__((vector_size(16)));   // matches async-LDS builtin param

#define HID    1024
#define GATES  4096
#define BATCH  32
#define TLEN   512
#define MROWS  (BATCH * TLEN)   // 16384

// ---------------- workspace layout (bytes) ----------------
#define OFF_XB     (0ull)
#define SZ_XB      ((size_t)MROWS * HID * 2)       // 32 MB  x in bf16
#define OFF_WIHB   (OFF_XB + SZ_XB)
#define SZ_WB      ((size_t)GATES * HID * 2)       // 8 MB
#define OFF_WHHB   (OFF_WIHB + SZ_WB)
#define OFF_XPROJ  (OFF_WHHB + SZ_WB)
#define SZ_XPROJ   ((size_t)MROWS * GATES * 4)     // 256 MB x_proj f32
#define OFF_HBUF   (OFF_XPROJ + SZ_XPROJ)
#define SZ_HBUF    ((size_t)2 * BATCH * HID * 2)   // 128 KB (double buffered h, bf16)
#define OFF_CNT    (OFF_HBUF + SZ_HBUF)

#define NWG2   128   // persistent workgroups for the scan
#define COLS   32    // h columns owned per WG
#define LROWS  128   // 4 gates * COLS weight rows in LDS

// ---------------- CDNA5 async global->LDS copy (ASYNCcnt path) ----------------
#if __has_builtin(__builtin_amdgcn_global_load_async_to_lds_b128) && \
    __has_builtin(__builtin_amdgcn_s_wait_asynccnt)
#define HAS_ASYNC_LDS 1
#else
#define HAS_ASYNC_LDS 0
#endif

__device__ __forceinline__ void copy16_g2l(const __bf16* gsrc, __bf16* ldst) {
#if HAS_ASYNC_LDS
    __builtin_amdgcn_global_load_async_to_lds_b128(
        (v4i*)(void*)gsrc, (v4i*)(void*)ldst, 0, 0);
#else
    *(v8bf*)ldst = *(const v8bf*)gsrc;
#endif
}

__device__ __forceinline__ void async_copy_fence() {
#if HAS_ASYNC_LDS
    __builtin_amdgcn_s_wait_asynccnt(0);
#endif
}

// =====================================================================
// f32 -> bf16 conversion (elementwise, grid-stride)
// =====================================================================
__global__ void f32_to_bf16_kernel(const float* __restrict__ src,
                                   __bf16* __restrict__ dst, int n) {
    int i = blockIdx.x * blockDim.x + threadIdx.x;
    int stride = gridDim.x * blockDim.x;
    for (; i < n; i += stride) dst[i] = (__bf16)src[i];
}

// zero h double-buffer + grid barrier counter (runs every launch -> replay safe)
__global__ void init_state_kernel(__bf16* __restrict__ hbuf,
                                  unsigned* __restrict__ cnt) {
    int i = blockIdx.x * blockDim.x + threadIdx.x;
    if (i == 0) *cnt = 0u;
    int n = 2 * BATCH * HID;
    int stride = gridDim.x * blockDim.x;
    for (; i < n; i += stride) hbuf[i] = (__bf16)0.0f;
}

// =====================================================================
// Phase 1: x_proj[M,4096] = x_bf16[M,1024] @ W_ih^T + (b_ih + b_hh)
// block = 256 thr (8 waves, 4(M)x2(N)), block tile 128x128, wave tile 32x64
// =====================================================================
__global__ void __launch_bounds__(256)
xproj_gemm_kernel(const __bf16* __restrict__ A,   // [MROWS, HID] row-major
                  const __bf16* __restrict__ W,   // [GATES, HID] row-major
                  const float*  __restrict__ bih,
                  const float*  __restrict__ bhh,
                  float* __restrict__ C)          // [MROWS, GATES]
{
    const int lane  = threadIdx.x & 31;
    const int wave  = threadIdx.x >> 5;
    const int waveM = wave & 3;
    const int waveN = wave >> 2;
    const int m0 = blockIdx.x * 128 + waveM * 32;
    const int n0 = blockIdx.y * 128 + waveN * 64;

    const int row  = lane & 15;
    const int aoff = (lane < 16) ? 0 : 16;   // A frag: K halves per half-wave
    const int boff = (lane < 16) ? 0 : 16;   // B frag: K halves per half-wave

    v8f acc[2][4] = {};

    for (int k = 0; k < HID; k += 32) {
        v16bf a[2], b[4];
#pragma unroll
        for (int mi = 0; mi < 2; ++mi) {
            const __bf16* p = A + (size_t)(m0 + mi * 16 + row) * HID + k + aoff;
            union { v16bf v; v8bf h[2]; } u;
            u.h[0] = *(const v8bf*)p;
            u.h[1] = *(const v8bf*)(p + 8);
            a[mi] = u.v;
            __builtin_prefetch(p + 32, 0, 3);   // near-cache prefetch, next K chunk
        }
#pragma unroll
        for (int ni = 0; ni < 4; ++ni) {
            const __bf16* p = W + (size_t)(n0 + ni * 16 + row) * HID + k + boff;
            b[ni] = *(const v16bf*)p;           // 32B contiguous (2x b128)
        }
#pragma unroll
        for (int mi = 0; mi < 2; ++mi)
#pragma unroll
            for (int ni = 0; ni < 4; ++ni)
                acc[mi][ni] = __builtin_amdgcn_wmma_f32_16x16x32_bf16(
                    false, a[mi], false, b[ni], (short)0, acc[mi][ni], false, false);
    }

    // C/D layout: VGPR i holds M=i (lanes 0-15) and M=8+i (lanes 16-31), col = lane%16
#pragma unroll
    for (int mi = 0; mi < 2; ++mi)
#pragma unroll
        for (int ni = 0; ni < 4; ++ni) {
            const int col  = n0 + ni * 16 + (lane & 15);
            const float bias = bih[col] + bhh[col];
            const int rbase = m0 + mi * 16 + ((lane >> 4) << 3);
            float* out = C + (size_t)rbase * GATES + col;
#pragma unroll
            for (int i = 0; i < 8; ++i)
                out[(size_t)i * GATES] = acc[mi][ni][i] + bias;
        }
}

// =====================================================================
// Phase 2: persistent LSTM scan. 128 WGs, WG g owns h cols [g*32, g*32+32).
// LDS: Wl[128][1024] bf16 (256KB, resident) + hs[32][1024] bf16 (64KB staging,
// aliased by gsm[32][128] f32 after the GEMM) = 320KB total.
// Staging uses GLOBAL_LOAD_ASYNC_TO_LDS (ASYNCcnt) when available.
// =====================================================================
__global__ void __launch_bounds__(256)
lstm_scan_kernel(const __bf16* __restrict__ Whh,    // [GATES, HID] bf16
                 const float*  __restrict__ xproj,  // [MROWS, GATES]
                 __bf16* __restrict__ hbuf,         // [2][BATCH][HID] bf16
                 float* __restrict__ out,           // [BATCH, TLEN, HID]
                 unsigned* __restrict__ cnt)
{
    extern __shared__ char smem[];
    __bf16* Wl  = (__bf16*)smem;             // 256 KB resident weight slice
    __bf16* hs  = (__bf16*)(smem + 262144);  // 64 KB h staging
    float*  gsm = (float*)(smem + 262144);   // 16 KB gate exchange (aliases hs)

    const int g    = blockIdx.x;
    const int c0   = g * COLS;
    const int tid  = threadIdx.x;
    const int lane = tid & 31;
    const int wave = tid >> 5;

    // ---- preload W_hh slice into LDS (async DMA, once) ----
    // local row r (0..127): gate q=r>>5, col j=r&31 -> global row q*1024 + c0 + j
    {
        const int r = tid >> 1;
        const int half = tid & 1;
        const int q = r >> 5, j = r & 31;
        const __bf16* src = Whh + (size_t)(q * HID + c0 + j) * HID + half * 512;
        __bf16* dst = Wl + (size_t)r * HID + half * 512;
        for (int u = 0; u < 512; u += 8)
            copy16_g2l(src + u, dst + u);
        async_copy_fence();
    }
    float creg[4] = {0.f, 0.f, 0.f, 0.f};   // cell state, 4 (b,j) pairs / thread
    __syncthreads();

    const int mtile  = wave & 1;             // 2 M tiles (batch 0-15 / 16-31)
    const int ntile0 = (wave >> 1) * 2;      // each wave: 2 of 8 N tiles
    const int aoff = (lane < 16) ? 0 : 16;
    const int boff = (lane < 16) ? 0 : 16;
    const int arow  = mtile * 16 + (lane & 15);
    const int brow0 = (ntile0 + 0) * 16 + (lane & 15);
    const int brow1 = (ntile0 + 1) * 16 + (lane & 15);

    for (int t = 0; t < TLEN; ++t) {
        // ---- stage h(t) into LDS: 256 thr x 256B, async global->LDS ----
        {
            const __bf16* hsrc = hbuf + (size_t)(t & 1) * BATCH * HID + tid * 128;
            __bf16* hdst = hs + tid * 128;
#pragma unroll
            for (int u = 0; u < 128; u += 8)
                copy16_g2l(hsrc + u, hdst + u);
            async_copy_fence();
        }
        __syncthreads();

        // ---- gates_raw[32,128] = h[32,1024] @ Wl[128,1024]^T ----
        v8f acc0 = {}, acc1 = {};
        for (int k = 0; k < HID; k += 32) {
            union { v16bf v; v8bf h[2]; } ua;
            ua.h[0] = *(const v8bf*)(hs + (size_t)arow * HID + k + aoff);
            ua.h[1] = *(const v8bf*)(hs + (size_t)arow * HID + k + aoff + 8);
            v16bf b0 = *(const v16bf*)(Wl + (size_t)brow0 * HID + k + boff);
            v16bf b1 = *(const v16bf*)(Wl + (size_t)brow1 * HID + k + boff);
            acc0 = __builtin_amdgcn_wmma_f32_16x16x32_bf16(
                false, ua.v, false, b0, (short)0, acc0, false, false);
            acc1 = __builtin_amdgcn_wmma_f32_16x16x32_bf16(
                false, ua.v, false, b1, (short)0, acc1, false, false);
        }
        __syncthreads();   // done reading hs -> safe to alias with gsm

        // ---- scatter accumulators into gate-exchange LDS ----
        {
            const int b  = mtile * 16 + ((lane >> 4) << 3);
            const int n0_ = ntile0 * 16 + (lane & 15);
#pragma unroll
            for (int i = 0; i < 8; ++i) {
                gsm[(b + i) * LROWS + n0_]      = acc0[i];
                gsm[(b + i) * LROWS + n0_ + 16] = acc1[i];
            }
        }
        __syncthreads();

        // ---- elementwise LSTM cell: 1024 (b,j) pairs, 4 per thread ----
#pragma unroll
        for (int r = 0; r < 4; ++r) {
            const int p = tid + 256 * r;
            const int b = p >> 5;
            const int j = p & 31;
            const float* xp = xproj + ((size_t)b * TLEN + t) * GATES + c0 + j;
            float gi = gsm[b * LROWS +  0 + j] + xp[0];
            float gf = gsm[b * LROWS + 32 + j] + xp[1024];
            float gg = gsm[b * LROWS + 64 + j] + xp[2048];
            float go = gsm[b * LROWS + 96 + j] + xp[3072];
            gi = 1.f / (1.f + __expf(-gi));
            gf = 1.f / (1.f + __expf(-gf));
            gg = tanhf(gg);
            go = 1.f / (1.f + __expf(-go));
            const float c = gf * creg[r] + gi * gg;
            creg[r] = c;
            const float h = go * tanhf(c);
            out[((size_t)b * TLEN + t) * HID + c0 + j] = h;
            hbuf[(size_t)((t + 1) & 1) * BATCH * HID + (size_t)b * HID + c0 + j] = (__bf16)h;
        }

        // ---- device-scope grid barrier ----
        __threadfence();
        __syncthreads();
        if (tid == 0) {
            __hip_atomic_fetch_add(cnt, 1u, __ATOMIC_ACQ_REL, __HIP_MEMORY_SCOPE_AGENT);
            const unsigned target = (unsigned)(t + 1) * (unsigned)gridDim.x;
            while (__hip_atomic_load(cnt, __ATOMIC_ACQUIRE, __HIP_MEMORY_SCOPE_AGENT) < target)
                __builtin_amdgcn_s_sleep(1);
        }
        __syncthreads();
    }
}

// =====================================================================
extern "C" void kernel_launch(void* const* d_in, const int* in_sizes, int n_in,
                              void* d_out, int out_size, void* d_ws, size_t ws_size,
                              hipStream_t stream) {
    (void)in_sizes; (void)n_in; (void)out_size; (void)ws_size;
    const float* x   = (const float*)d_in[0];   // [32,512,1024]
    const float* Wih = (const float*)d_in[1];   // [4096,1024]
    const float* Whh = (const float*)d_in[2];   // [4096,1024]
    const float* bih = (const float*)d_in[3];   // [4096]
    const float* bhh = (const float*)d_in[4];   // [4096]
    float* out = (float*)d_out;                 // [32,512,1024]

    char* ws = (char*)d_ws;
    __bf16*   xb    = (__bf16*)(ws + OFF_XB);
    __bf16*   wihb  = (__bf16*)(ws + OFF_WIHB);
    __bf16*   whhb  = (__bf16*)(ws + OFF_WHHB);
    float*    xproj = (float*)(ws + OFF_XPROJ);
    __bf16*   hbuf  = (__bf16*)(ws + OFF_HBUF);
    unsigned* cnt   = (unsigned*)(ws + OFF_CNT);

    // 1) precision conversion
    f32_to_bf16_kernel<<<2048, 256, 0, stream>>>(x,   xb,   MROWS * HID);
    f32_to_bf16_kernel<<<1024, 256, 0, stream>>>(Wih, wihb, GATES * HID);
    f32_to_bf16_kernel<<<1024, 256, 0, stream>>>(Whh, whhb, GATES * HID);

    // 2) reset h state + grid-barrier counter (every call -> replay-safe)
    init_state_kernel<<<256, 256, 0, stream>>>(hbuf, cnt);

    // 3) big input-projection GEMM via WMMA
    dim3 g1(MROWS / 128, GATES / 128);
    xproj_gemm_kernel<<<g1, 256, 0, stream>>>(xb, wihb, bih, bhh, xproj);

    // 4) persistent recurrent scan, 320KB LDS per WG
    (void)hipFuncSetAttribute((const void*)lstm_scan_kernel,
                              hipFuncAttributeMaxDynamicSharedMemorySize, 327680);
    lstm_scan_kernel<<<NWG2, 256, 327680, stream>>>(whhb, xproj, hbuf, out, cnt);
}
